// WeightMaskMultiHeadAttention_13022340841949
// MI455X (gfx1250) — compile-verified, matrix operand order checked by reading, not run on hardware
//
#include <hip/hip_runtime.h>
#include <hip/hip_bf16.h>
#include <math.h>

typedef __attribute__((ext_vector_type(16))) __bf16 v16bf;
typedef __attribute__((ext_vector_type(8)))  __bf16 v8bf;
typedef __attribute__((ext_vector_type(4)))  __bf16 v4bf;
typedef __attribute__((ext_vector_type(2)))  __bf16 v2bf;
typedef __attribute__((ext_vector_type(8)))  float  v8f;
typedef unsigned int u32x4 __attribute__((ext_vector_type(4)));
typedef int          i32x4 __attribute__((ext_vector_type(4)));
typedef int          i32x8 __attribute__((ext_vector_type(8)));

#define WMMA_BF16(a, b, c) \
  __builtin_amdgcn_wmma_f32_16x16x32_bf16(false, (a), false, (b), (short)0, (c), false, false)

__device__ __forceinline__ uint32_t lds_addr_of(const void* p) {
    // generic LDS pointers carry the LDS byte offset in the low 32 bits
    return (uint32_t)(uintptr_t)p;
}

__device__ __forceinline__ void async_copy_b128(uint32_t lds, const void* gptr) {
    asm volatile("global_load_async_to_lds_b128 %0, %1, off"
                 :: "v"(lds), "v"((uint64_t)(uintptr_t)gptr) : "memory");
}
__device__ __forceinline__ void wait_async0() {
    asm volatile("s_wait_asynccnt 0" ::: "memory");
}

// ---------------------------------------------------------------------------
// fp32 -> bf16 elementwise convert (x and activations)
// ---------------------------------------------------------------------------
__global__ void convert_bf16_kernel(const float* __restrict__ src,
                                    __bf16* __restrict__ dst, int n4) {
    int id = blockIdx.x * blockDim.x + threadIdx.x;
    if (id >= n4) return;
    float4 f = ((const float4*)src)[id];
    v4bf t;
    t[0] = (__bf16)f.x; t[1] = (__bf16)f.y;
    t[2] = (__bf16)f.z; t[3] = (__bf16)f.w;
    ((v4bf*)dst)[id] = t;
}

// ---------------------------------------------------------------------------
// fp32 W[K=1024][N=1024] -> bf16 WT[N][K] (transpose + convert, one-shot)
// ---------------------------------------------------------------------------
__global__ void transpose_convert_kernel(const float* __restrict__ W,
                                         __bf16* __restrict__ WT) {
    int id = blockIdx.x * blockDim.x + threadIdx.x;   // 1024 * 256 threads
    int n  = id & 1023;
    int k0 = (id >> 10) * 4;
    v4bf t;
    #pragma unroll
    for (int j = 0; j < 4; ++j) t[j] = (__bf16)W[(size_t)(k0 + j) * 1024 + n];
    *(v4bf*)(WT + (size_t)n * 1024 + k0) = t;
}

// ---------------------------------------------------------------------------
// 14x14 average pool of weight_mask -> wm[32][257] (CLS token = 1)
// ---------------------------------------------------------------------------
__global__ void mask_pool_kernel(const float* __restrict__ wmask,
                                 float* __restrict__ wm) {
    int idx = blockIdx.x * blockDim.x + threadIdx.x;
    const int TOT = 32 * 257;
    if (idx >= TOT) return;
    int bt = idx / 257;
    int j  = idx % 257;
    if (j == 0) { wm[idx] = 1.0f; return; }
    int p  = j - 1;
    int gy = p >> 4;
    int gx = p & 15;
    const float* base = wmask + (size_t)bt * 224 * 224;
    float s = 0.0f;
    for (int r = 0; r < 14; ++r) {
        const float* row = base + (size_t)(gy * 14 + r) * 224 + gx * 14;
        #pragma unroll
        for (int c = 0; c < 14; ++c) s += row[c];
    }
    wm[idx] = s * (1.0f / 196.0f);
}

// ---------------------------------------------------------------------------
// GEMM: C[M,1024] = A_bf16[M,1024] @ WT_bf16[1024(N),1024(K)]^T + bias.
// Block 256 threads (8 waves), tile 128x64, BK=32; wave computes 32x32
// (2 A-frags x 2 B-frags -> 4 WMMA per k-step). Both LDS tiles live in WMMA
// fragment layout; since data is bf16 and contiguous along K on both sides,
// staging is pure per-lane async b128 copies (GLOBAL_LOAD_ASYNC_TO_LDS_B128,
// ASYNCcnt). Fragment reads are contiguous v16bf (2x ds_load_b128).
// ---------------------------------------------------------------------------
template <bool OUT_BF16>
__global__ __launch_bounds__(256)
void gemm_bf16_kernel(const __bf16* __restrict__ A, const __bf16* __restrict__ WT,
                      const float* __restrict__ bias, void* __restrict__ Cv,
                      int M) {
    const int K = 1024, N = 1024;
    __shared__ v16bf AsF[8][32];   // 8 m-subtiles (128 rows)
    __shared__ v16bf BsF[4][32];   // 4 n-groups  (64 cols)

    int tid  = threadIdx.x;
    int lane = tid & 31;
    int wave = tid >> 5;
    int wm_  = wave & 3;          // 32-row quadrant
    int wn_  = wave >> 2;         // 32-col half
    int half = lane >> 4;
    int lr   = lane & 15;

    int m0 = blockIdx.y * 128;
    int n0 = blockIdx.x * 64;

    // A staging: thread -> row ar (0..127), octet pair {ao, ao+16}
    int ar = tid >> 1;
    int ao = (tid & 1) * 8;
    int agr = m0 + ar; if (agr >= M) agr = M - 1;   // clamp (rows >= M discarded)
    const __bf16* a_src = A + (size_t)agr * K + ao;
    uint32_t a_lds = lds_addr_of(
        (__bf16*)&AsF[ar >> 4][(ar & 15) + (((ao >> 3) & 1) << 4)]);

    // B staging: thread -> col bc (0..63), k-octet bo
    int bc = tid & 63;
    int bo = (tid >> 6) * 8;
    const __bf16* b_src = WT + (size_t)(n0 + bc) * K + bo;
    uint32_t b_lds = lds_addr_of(
        (__bf16*)&BsF[bc >> 4][(bc & 15) + (((bo >> 3) & 1) << 4)] + (((bo >> 4) & 1) << 3));

    v8f acc00 = {}, acc01 = {}, acc10 = {}, acc11 = {};

    for (int k0 = 0; k0 < K; k0 += 32) {
        __syncthreads();
        async_copy_b128(a_lds,      a_src + k0);        // octet ao      -> e0=0
        async_copy_b128(a_lds + 16, a_src + k0 + 16);   // octet ao+16   -> e0=8
        async_copy_b128(b_lds,      b_src + k0);
        wait_async0();
        __syncthreads();

        v16bf a0 = AsF[wm_ * 2 + 0][lane];
        v16bf a1 = AsF[wm_ * 2 + 1][lane];
        v16bf b0 = BsF[wn_ * 2 + 0][lane];
        v16bf b1 = BsF[wn_ * 2 + 1][lane];
        acc00 = WMMA_BF16(a0, b0, acc00);
        acc01 = WMMA_BF16(a0, b1, acc01);
        acc10 = WMMA_BF16(a1, b0, acc10);
        acc11 = WMMA_BF16(a1, b1, acc11);
    }

    int c0 = n0 + wn_ * 32 + lr;
    int c1 = c0 + 16;
    float bias0 = bias[c0], bias1 = bias[c1];
    #pragma unroll
    for (int v = 0; v < 8; ++v) {
        int r0 = m0 + wm_ * 32 + v + 8 * half;
        int r1 = r0 + 16;
        if (OUT_BF16) {
            __bf16* C = (__bf16*)Cv;
            if (r0 < M) {
                C[(size_t)r0 * N + c0] = (__bf16)(acc00[v] + bias0);
                C[(size_t)r0 * N + c1] = (__bf16)(acc01[v] + bias1);
            }
            if (r1 < M) {
                C[(size_t)r1 * N + c0] = (__bf16)(acc10[v] + bias0);
                C[(size_t)r1 * N + c1] = (__bf16)(acc11[v] + bias1);
            }
        } else {
            float* C = (float*)Cv;
            if (r0 < M) {
                C[(size_t)r0 * N + c0] = acc00[v] + bias0;
                C[(size_t)r0 * N + c1] = acc01[v] + bias1;
            }
            if (r1 < M) {
                C[(size_t)r1 * N + c0] = acc10[v] + bias0;
                C[(size_t)r1 * N + c1] = acc11[v] + bias1;
            }
        }
    }
}

// ---------------------------------------------------------------------------
// Attention: one wave per (bt, head, 16-query tile). K panel loaded with the
// Tensor Data Mover (zero-pads rows >= 257); QK^T fragments read directly
// from the raw row panel (d is contiguous per lane). V re-staged into the
// same LDS buffer in B-fragment layout for P@V.
// ---------------------------------------------------------------------------
__global__ __launch_bounds__(32)
void attn_kernel(const __bf16* __restrict__ Q, const __bf16* __restrict__ Km,
                 const __bf16* __restrict__ V, const float* __restrict__ wm,
                 __bf16* __restrict__ O) {
    const int Nn = 257, D = 1024, HD = 64;
    __shared__ float sc[16][288];
    __shared__ float l_s[16];
    // K raw panel: 272 rows x 64 bf16 = 17408 elems; later reused as
    // V fragment panel: 9*4*32*16 = 18432 elems.
    __shared__ __align__(32) __bf16 panel[18432];

    int bid = blockIdx.x;
    int qt  = bid % 17;
    int bh  = bid / 17;
    int h   = bh & 15;
    int bt  = bh >> 4;

    int lane = threadIdx.x;
    int half = lane >> 4;
    int lr   = lane & 15;
    int kb   = half * 8;
    int q0   = qt * 16;

    size_t seqbase = (size_t)bt * Nn;
    const __bf16* Qb = Q  + seqbase * D + h * HD;
    const __bf16* Kb = Km + seqbase * D + h * HD;
    const __bf16* Vb = V  + seqbase * D + h * HD;

    // ---- TDM: load K panel [272 x 64] bf16 into LDS (rows >=257 zero) ----
    {
        uint32_t lds = lds_addr_of(&panel[0]);
        uint64_t ga  = (uint64_t)(uintptr_t)Kb;
        u32x4 g0;
        g0[0] = 1u;                                   // count=1, user mode
        g0[1] = lds;                                  // lds_addr
        g0[2] = (uint32_t)ga;                         // global_addr[31:0]
        g0[3] = (uint32_t)((ga >> 32) & 0x1FFFFFFu) | (2u << 30);  // addr[56:32], type=2
        i32x8 g1;
        g1[0] = 1 << 16;                              // data_size = 1 (2 bytes)
        g1[1] = (int)(64u << 16);                     // tensor_dim0 = 64
        g1[2] = (int)(257u << 16);                    // tensor_dim1 = 257
        g1[3] = (int)(64u << 16);                     // tile_dim0 = 64
        g1[4] = 272;                                  // tile_dim1 = 272
        g1[5] = 1024;                                 // tensor_dim0_stride = 1024
        g1[6] = 0;
        g1[7] = 0;
        i32x4 gz4 = {0, 0, 0, 0};
        i32x8 gz8 = {0, 0, 0, 0, 0, 0, 0, 0};
        __builtin_amdgcn_tensor_load_to_lds(g0, g1, gz4, gz4, gz8, 0);
    }

    // ---- Q fragments while TDM runs ----
    v16bf aq0, aq1;
    {
        int q  = q0 + lr;
        bool ok = q < Nn;
        const __bf16* qp = Qb + (size_t)(ok ? q : 0) * D;
        v8bf l0 = *(const v8bf*)(qp + kb);
        v8bf h0 = *(const v8bf*)(qp + 16 + kb);
        v8bf l1 = *(const v8bf*)(qp + 32 + kb);
        v8bf h1 = *(const v8bf*)(qp + 48 + kb);
        aq0 = __builtin_shufflevector(l0, h0, 0,1,2,3,4,5,6,7,8,9,10,11,12,13,14,15);
        aq1 = __builtin_shufflevector(l1, h1, 0,1,2,3,4,5,6,7,8,9,10,11,12,13,14,15);
        if (!ok) { v16bf z = {}; aq0 = z; aq1 = z; }
    }
    float wmq[8];
    #pragma unroll
    for (int v = 0; v < 8; ++v) {
        int q = q0 + v + 8 * half;
        wmq[v] = (q < Nn) ? wm[seqbase + q] : 0.0f;
    }

    __builtin_amdgcn_s_wait_tensorcnt(0);
    __syncthreads();

    // ---- pass 1: scores into LDS (fragments straight from raw K rows) ----
    for (int kt = 0; kt < 17; ++kt) {
        int k0  = kt * 16;
        int key = k0 + lr;
        const __bf16* kr = &panel[key * 64];
        v8bf l0 = *(const v8bf*)(kr + kb);
        v8bf h0 = *(const v8bf*)(kr + 16 + kb);
        v8bf l1 = *(const v8bf*)(kr + 32 + kb);
        v8bf h1 = *(const v8bf*)(kr + 48 + kb);
        v16bf bk0 = __builtin_shufflevector(l0, h0, 0,1,2,3,4,5,6,7,8,9,10,11,12,13,14,15);
        v16bf bk1 = __builtin_shufflevector(l1, h1, 0,1,2,3,4,5,6,7,8,9,10,11,12,13,14,15);
        v8f s = {};
        s = WMMA_BF16(aq0, bk0, s);
        s = WMMA_BF16(aq1, bk1, s);
        bool ok   = key < Nn;
        float wmk = ok ? wm[seqbase + key] : 0.0f;
        #pragma unroll
        for (int v = 0; v < 8; ++v) {
            float val = ok ? s[v] * 0.125f * wmq[v] * wmk : -1.0e30f;
            sc[v + 8 * half][k0 + lr] = val;
        }
    }
    for (int idx = lane; idx < 256; idx += 32)
        sc[idx >> 4][272 + (idx & 15)] = -1.0e30f;
    __syncthreads();

    // ---- softmax stats: lanes (r, r+16) share row r, 144 cols each ----
    {
        float4* rowp = (float4*)&sc[lr][half * 144];
        float mloc = -3.0e38f;
        for (int j = 0; j < 36; ++j) {
            float4 v = rowp[j];
            mloc = fmaxf(mloc, fmaxf(fmaxf(v.x, v.y), fmaxf(v.z, v.w)));
        }
        float mrow = fmaxf(mloc, __shfl_xor(mloc, 16, 32));
        float ssum = 0.0f;
        for (int j = 0; j < 36; ++j) {
            float4 v = rowp[j];
            v.x = __expf(v.x - mrow);
            v.y = __expf(v.y - mrow);
            v.z = __expf(v.z - mrow);
            v.w = __expf(v.w - mrow);
            rowp[j] = v;
            ssum += v.x + v.y + v.z + v.w;
        }
        float tot = ssum + __shfl_xor(ssum, 16, 32);
        if (half == 0) l_s[lr] = tot;
    }
    __syncthreads();

    // ---- stage V into panel in B-fragment layout ----
    // element (key, d): frag = (key>>5)*4 + (d>>4); lane' = (d&15)+16*((key>>3)&1)
    //                   e = (key&7) + 8*((key>>4)&1)
    {
        int c = lane * 2;
        int i = c >> 4;
        for (int r = 0; r < Nn; ++r) {
            v2bf f = *(const v2bf*)(Vb + (size_t)r * D + c);
            int kc = r >> 5, kl = r & 31;
            int e  = (kl & 7) + (((kl >> 4) & 1) << 3);
            int lA = (c & 15) + (((kl >> 3) & 1) << 4);
            size_t fb = (size_t)((kc * 4 + i) * 32) * 16 + e;
            panel[fb + (size_t)lA * 16]       = f[0];
            panel[fb + (size_t)(lA + 1) * 16] = f[1];
        }
        __bf16 z = (__bf16)0.0f;
        for (int r = Nn; r < 288; ++r) {
            int kc = r >> 5, kl = r & 31;
            int e  = (kl & 7) + (((kl >> 4) & 1) << 3);
            int lA = (c & 15) + (((kl >> 3) & 1) << 4);
            size_t fb = (size_t)((kc * 4 + i) * 32) * 16 + e;
            panel[fb + (size_t)lA * 16]       = z;
            panel[fb + (size_t)(lA + 1) * 16] = z;
        }
    }
    __syncthreads();

    // ---- pass 2: O = P @ V ----
    const v16bf* fragp = (const v16bf*)panel;
    v8f o[4] = {{}, {}, {}, {}};
    for (int kc = 0; kc < 9; ++kc) {
        v16bf ap;
        {
            const float4* p0 = (const float4*)&sc[lr][kc * 32 + kb];
            const float4* p1 = (const float4*)&sc[lr][kc * 32 + 16 + kb];
            float4 f0 = p0[0], f1 = p0[1];
            float4 f2 = p1[0], f3 = p1[1];
            ap[0]  = (__bf16)f0.x; ap[1]  = (__bf16)f0.y;
            ap[2]  = (__bf16)f0.z; ap[3]  = (__bf16)f0.w;
            ap[4]  = (__bf16)f1.x; ap[5]  = (__bf16)f1.y;
            ap[6]  = (__bf16)f1.z; ap[7]  = (__bf16)f1.w;
            ap[8]  = (__bf16)f2.x; ap[9]  = (__bf16)f2.y;
            ap[10] = (__bf16)f2.z; ap[11] = (__bf16)f2.w;
            ap[12] = (__bf16)f3.x; ap[13] = (__bf16)f3.y;
            ap[14] = (__bf16)f3.z; ap[15] = (__bf16)f3.w;
        }
        #pragma unroll
        for (int i = 0; i < 4; ++i) {
            v16bf bv = fragp[(kc * 4 + i) * 32 + lane];
            o[i] = WMMA_BF16(ap, bv, o[i]);
        }
    }

    // ---- normalize + store (bf16 for the Wo GEMM) ----
    #pragma unroll
    for (int v = 0; v < 8; ++v) {
        int q = q0 + v + 8 * half;
        if (q < Nn) {
            float inv = 1.0f / l_s[v + 8 * half];
            __bf16* op = O + (seqbase + q) * D + h * HD;
            op[lr]      = (__bf16)(o[0][v] * inv);
            op[16 + lr] = (__bf16)(o[1][v] * inv);
            op[32 + lr] = (__bf16)(o[2][v] * inv);
            op[48 + lr] = (__bf16)(o[3][v] * inv);
        }
    }
}

// ---------------------------------------------------------------------------
extern "C" void kernel_launch(void* const* d_in, const int* in_sizes, int n_in,
                              void* d_out, int out_size, void* d_ws, size_t ws_size,
                              hipStream_t stream) {
    const float* x     = (const float*)d_in[0];
    const float* wmask = (const float*)d_in[1];
    const float* Wq    = (const float*)d_in[2];
    const float* bq    = (const float*)d_in[3];
    const float* Wk    = (const float*)d_in[4];
    const float* bk    = (const float*)d_in[5];
    const float* Wv    = (const float*)d_in[6];
    const float* bv    = (const float*)d_in[7];
    const float* Wo    = (const float*)d_in[8];
    const float* bo    = (const float*)d_in[9];
    float* out = (float*)d_out;

    const int TOK = 32 * 257;                 // 8224 tokens
    char* ws = (char*)d_ws;
    size_t off = 0;
    auto alloc = [&](size_t bytes) {
        char* p = ws + off;
        off = (off + bytes + 255) & ~(size_t)255;
        return p;
    };
    float*  wm  = (float*)alloc((size_t)TOK * 4);
    __bf16* xbf = (__bf16*)alloc((size_t)TOK * 1024 * 2);
    __bf16* WqT = (__bf16*)alloc((size_t)1024 * 1024 * 2);
    __bf16* WkT = (__bf16*)alloc((size_t)1024 * 1024 * 2);
    __bf16* WvT = (__bf16*)alloc((size_t)1024 * 1024 * 2);
    __bf16* WoT = (__bf16*)alloc((size_t)1024 * 1024 * 2);
    __bf16* Qb  = (__bf16*)alloc((size_t)TOK * 1024 * 2);
    __bf16* Kb  = (__bf16*)alloc((size_t)TOK * 1024 * 2);
    __bf16* Vb  = (__bf16*)alloc((size_t)TOK * 1024 * 2);
    __bf16* Ab  = (__bf16*)alloc((size_t)TOK * 1024 * 2);

    int n4 = TOK * 1024 / 4;
    convert_bf16_kernel<<<(n4 + 255) / 256, 256, 0, stream>>>(x, xbf, n4);
    transpose_convert_kernel<<<1024, 256, 0, stream>>>(Wq, WqT);
    transpose_convert_kernel<<<1024, 256, 0, stream>>>(Wk, WkT);
    transpose_convert_kernel<<<1024, 256, 0, stream>>>(Wv, WvT);
    transpose_convert_kernel<<<1024, 256, 0, stream>>>(Wo, WoT);
    mask_pool_kernel<<<(TOK + 255) / 256, 256, 0, stream>>>(wmask, wm);

    dim3 gg(16, (TOK + 127) / 128);           // 16 x 65
    gemm_bf16_kernel<true><<<gg, 256, 0, stream>>>(xbf, WqT, bq, Qb, TOK);
    gemm_bf16_kernel<true><<<gg, 256, 0, stream>>>(xbf, WkT, bk, Kb, TOK);
    gemm_bf16_kernel<true><<<gg, 256, 0, stream>>>(xbf, WvT, bv, Vb, TOK);

    attn_kernel<<<32 * 16 * 17, 32, 0, stream>>>(Qb, Kb, Vb, wm, Ab);

    gemm_bf16_kernel<false><<<gg, 256, 0, stream>>>(Ab, WoT, bo, out, TOK);
}
